// MPSet2Set_54039278518616
// MI455X (gfx1250) — matrix operand order repaired and checked by reading, not compiled, not run on hardware
//
#include <hip/hip_runtime.h>

typedef unsigned short u16;
typedef __attribute__((ext_vector_type(16))) __bf16 v16bf;
typedef __attribute__((ext_vector_type(8)))  float  v8f;

union FragBF { v16bf f; uint4 q[2]; };

__device__ __forceinline__ u16 f2bf(float x) {
    unsigned int u = __float_as_uint(x);
    unsigned int r = u + 0x7FFFu + ((u >> 16) & 1u);
    return (u16)(r >> 16);
}
__device__ __forceinline__ float bf2f(u16 u) { return __uint_as_float(((unsigned)u) << 16); }

__device__ __forceinline__ v8f wmma_bf(const FragBF& a, const FragBF& b, v8f c) {
    return __builtin_amdgcn_wmma_f32_16x16x32_bf16(false, a.f, false, b.f, (short)0, c, false, false);
}

// ---------------- constants / workspace layout ----------------
#define BB 128
#define NN 512
#define HH 256
#define FJ 8
#define GINP 800          // 776 padded to 25*32
#define NEGV (-1e9f)

static constexpr size_t SZ_BNH = (size_t)BB * NN * HH * 2;       // bf16 [B,N,H]
static constexpr size_t OFF_HBF  = 0;
static constexpr size_t OFF_QBF  = OFF_HBF  + SZ_BNH;            // later reused as r*h
static constexpr size_t OFF_MT   = OFF_QBF  + SZ_BNH;            // mT [B,H,N]; later reused as z
static constexpr size_t OFF_MSG  = OFF_MT   + SZ_BNH;
static constexpr size_t OFF_WAT  = OFF_MSG  + SZ_BNH;
static constexpr size_t OFF_WMT  = OFF_WAT  + (size_t)HH*HH*2;
static constexpr size_t OFF_WZT  = OFF_WMT  + (size_t)HH*HH*2;
static constexpr size_t OFF_WRT  = OFF_WZT  + (size_t)HH*GINP*2;
static constexpr size_t OFF_WHT  = OFF_WRT  + (size_t)HH*GINP*2;
static constexpr size_t OFF_HMEAN= OFF_WHT  + (size_t)HH*GINP*2;
static constexpr size_t OFF_JETS = OFF_HMEAN+ (size_t)BB*HH*2;

// ---------------- prep kernels ----------------
__global__ void k_prep_hh(const float* __restrict__ W, u16* __restrict__ WT) {
    int c = blockIdx.x;
    for (int k = threadIdx.x; k < HH; k += blockDim.x)
        WT[c * HH + k] = f2bf(W[k * HH + c]);
}

__global__ void k_prep_gin(const float* __restrict__ W, u16* __restrict__ WT) {
    int c = blockIdx.x;
    for (int k = threadIdx.x; k < GINP; k += blockDim.x)
        WT[c * GINP + k] = (k < 776) ? f2bf(W[k * HH + c]) : (u16)0;
}

__global__ void k_mean_jets(const float* __restrict__ h, const float* __restrict__ jets,
                            u16* __restrict__ hmean, u16* __restrict__ jetsb) {
    int b = blockIdx.x, d = threadIdx.x;
    float s = 0.f;
    for (int n = 0; n < NN; ++n) s += h[((size_t)(b * NN + n)) * HH + d];
    hmean[b * HH + d] = f2bf(s * (1.0f / (float)NN));
    if (d < FJ) jetsb[b * FJ + d] = f2bf(jets[b * FJ + d]);
}

// ---------------- K1: h->bf16, q = h@Wa+ba, mT = (h@Wm+bm)^T  (32 rows/wave) ----------------
__global__ __launch_bounds__(32)
void k1_qm(const float* __restrict__ h, const float* __restrict__ ba, const float* __restrict__ bmv,
           const u16* __restrict__ WaT, const u16* __restrict__ WmT,
           u16* __restrict__ hbf, u16* __restrict__ qbf, u16* __restrict__ mTbf) {
    __shared__ u16 sh[32 * HH];                        // 16 KB
    int lane = threadIdx.x, blk = blockIdx.x;
    int b = blk >> 4, n0 = (blk & 15) << 5;
    const float* hrow = h + ((size_t)(b * NN + n0)) * HH;
    u16* hbrow = hbf + ((size_t)(b * NN + n0)) * HH;
    for (int i = lane; i < 32 * HH / 4; i += 32) {
        float4 v = ((const float4*)hrow)[i];
        ushort4 u = make_ushort4(f2bf(v.x), f2bf(v.y), f2bf(v.z), f2bf(v.w));
        ((ushort4*)sh)[i] = u;
        ((ushort4*)hbrow)[i] = u;
    }
    __syncthreads();
    int m = lane & 15, half = lane >> 4, cl = lane & 15;
    FragBF afr[2][8];
#pragma unroll
    for (int t = 0; t < 2; ++t)
#pragma unroll
        for (int kc = 0; kc < 8; ++kc) {
            const u16* p0 = sh + (t * 16 + m) * HH + kc * 32 + half * 8;
            afr[t][kc].q[0] = *(const uint4*)p0;
            afr[t][kc].q[1] = *(const uint4*)(p0 + 16);
        }
    for (int c2 = 0; c2 < 16; ++c2) {
        int col = c2 * 16 + cl;
        v8f aq0 = {}, aq1 = {}, am0 = {}, am1 = {};
#pragma unroll
        for (int kc = 0; kc < 8; ++kc) {
            FragBF bq, bm_;
            const u16* pq = WaT + col * HH + kc * 32 + half * 8;
            const u16* pm = WmT + col * HH + kc * 32 + half * 8;
            bq.q[0] = *(const uint4*)pq;  bq.q[1] = *(const uint4*)(pq + 16);
            bm_.q[0] = *(const uint4*)pm; bm_.q[1] = *(const uint4*)(pm + 16);
            aq0 = wmma_bf(afr[0][kc], bq, aq0);
            aq1 = wmma_bf(afr[1][kc], bq, aq1);
            am0 = wmma_bf(afr[0][kc], bm_, am0);
            am1 = wmma_bf(afr[1][kc], bm_, am1);
        }
        float biasq = ba[col], biasm = bmv[col];
        u16 mv0[8], mv1[8];
#pragma unroll
        for (int v = 0; v < 8; ++v) {
            int r0 = n0 + v + 8 * half, r1 = r0 + 16;
            qbf[((size_t)(b * NN + r0)) * HH + col] = f2bf(aq0[v] + biasq);
            qbf[((size_t)(b * NN + r1)) * HH + col] = f2bf(aq1[v] + biasq);
            mv0[v] = f2bf(am0[v] + biasm);
            mv1[v] = f2bf(am1[v] + biasm);
        }
        ushort4* d0 = (ushort4*)(mTbf + ((size_t)(b * HH + col)) * NN + n0 + 8 * half);
        d0[0] = make_ushort4(mv0[0], mv0[1], mv0[2], mv0[3]);
        d0[1] = make_ushort4(mv0[4], mv0[5], mv0[6], mv0[7]);
        ushort4* d1 = (ushort4*)(mTbf + ((size_t)(b * HH + col)) * NN + n0 + 16 + 8 * half);
        d1[0] = make_ushort4(mv1[0], mv1[1], mv1[2], mv1[3]);
        d1[1] = make_ushort4(mv1[4], mv1[5], mv1[6], mv1[7]);
    }
}

// ---------------- K2: fused logits -> softmax -> msg = tanh(A@m)  (32 rows/wave) ----------------
__global__ __launch_bounds__(32)
void k2_attn(const u16* __restrict__ qbf, const u16* __restrict__ hbf,
             const u16* __restrict__ mTbf, const float* __restrict__ mask,
             u16* __restrict__ msgbf) {
    __shared__ float slog[32 * NN];   // 64 KB, reused in place for bf16 A
    int lane = threadIdx.x, blk = blockIdx.x;
    int b = blk >> 4, n0 = (blk & 15) << 5;
    int m = lane & 15, half = lane >> 4, cl = lane & 15;
    const float* maskb = mask + b * NN;

    FragBF qa[2][8];
#pragma unroll
    for (int t = 0; t < 2; ++t) {
        const u16* qrow = qbf + ((size_t)(b * NN + n0 + t * 16)) * HH;
#pragma unroll
        for (int kc = 0; kc < 8; ++kc) {
            const u16* p0 = qrow + m * HH + kc * 32 + half * 8;
            qa[t][kc].q[0] = *(const uint4*)p0;
            qa[t][kc].q[1] = *(const uint4*)(p0 + 16);
        }
    }
    // scores: 32 rows x all 512 senders; one B-frag feeds both row tiles
    for (int n2 = 0; n2 < 32; ++n2) {
        int j = n2 * 16 + cl;
        const u16* hb = hbf + ((size_t)(b * NN + j)) * HH;
        v8f a0 = {}, a1 = {};
#pragma unroll
        for (int kc = 0; kc < 8; ++kc) {
            FragBF bb;
            const u16* p = hb + kc * 32 + half * 8;
            bb.q[0] = *(const uint4*)p; bb.q[1] = *(const uint4*)(p + 16);
            a0 = wmma_bf(qa[0][kc], bb, a0);
            a1 = wmma_bf(qa[1][kc], bb, a1);
        }
        float add = NEGV * (1.0f - maskb[j]);
#pragma unroll
        for (int v = 0; v < 8; ++v) {
            slog[(v + 8 * half) * NN + j]      = a0[v] + add;
            slog[(16 + v + 8 * half) * NN + j] = a1[v] + add;
        }
    }
    __syncthreads();
    // masked softmax, one lane per row, vectorized; bf16 conversion in place
    {
        int row = lane;
        float4* lr4 = (float4*)(slog + row * NN);
        float mx = -3.0e38f;
        for (int i = 0; i < 128; ++i) {
            float4 x = lr4[i];
            mx = fmaxf(mx, fmaxf(fmaxf(x.x, x.y), fmaxf(x.z, x.w)));
        }
        float sum = 0.f;
        for (int i = 0; i < 128; ++i) {
            float4 x = lr4[i];
            x.x = __expf(x.x - mx); x.y = __expf(x.y - mx);
            x.z = __expf(x.z - mx); x.w = __expf(x.w - mx);
            lr4[i] = x;
            sum += x.x + x.y + x.z + x.w;
        }
        float sc = maskb[n0 + row] / fmaxf(sum, 1e-20f);
        // pack 8 f32 -> 8 bf16 (one float4) per step; forward-safe in-place
        for (int i = 0; i < 64; ++i) {
            float4 x = lr4[2 * i], y = lr4[2 * i + 1];
            float4 p;
            p.x = __uint_as_float(((unsigned)f2bf(x.y * sc) << 16) | (unsigned)f2bf(x.x * sc));
            p.y = __uint_as_float(((unsigned)f2bf(x.w * sc) << 16) | (unsigned)f2bf(x.z * sc));
            p.z = __uint_as_float(((unsigned)f2bf(y.y * sc) << 16) | (unsigned)f2bf(y.x * sc));
            p.w = __uint_as_float(((unsigned)f2bf(y.w * sc) << 16) | (unsigned)f2bf(y.z * sc));
            lr4[i] = p;
        }
    }
    __syncthreads();
    // msg = tanh(A(32x512) @ m(512x256)); one B-frag feeds both row tiles
    const u16* sab = (const u16*)slog;          // row r at u16 offset r*1024
    for (int d2 = 0; d2 < 16; ++d2) {
        int col = d2 * 16 + cl;
        const u16* mcol = mTbf + ((size_t)(b * HH + col)) * NN;
        v8f c0 = {}, c1 = {};
#pragma unroll
        for (int mc = 0; mc < 16; ++mc) {
            FragBF a0, a1, bb;
            const u16* p0 = sab + (size_t)m * 1024 + mc * 32 + half * 8;
            const u16* p1 = sab + (size_t)(16 + m) * 1024 + mc * 32 + half * 8;
            a0.q[0] = *(const uint4*)p0; a0.q[1] = *(const uint4*)(p0 + 16);
            a1.q[0] = *(const uint4*)p1; a1.q[1] = *(const uint4*)(p1 + 16);
            const u16* p = mcol + mc * 32 + half * 8;
            bb.q[0] = *(const uint4*)p; bb.q[1] = *(const uint4*)(p + 16);
            c0 = wmma_bf(a0, bb, c0);
            c1 = wmma_bf(a1, bb, c1);
        }
#pragma unroll
        for (int v = 0; v < 8; ++v) {
            msgbf[((size_t)(b * NN + n0 + v + 8 * half)) * HH + col]      = f2bf(tanhf(c0[v]));
            msgbf[((size_t)(b * NN + n0 + 16 + v + 8 * half)) * HH + col] = f2bf(tanhf(c1[v]));
        }
    }
}

// ---------------- staging helper for GRU tiles (32 rows x 800) ----------------
__device__ __forceinline__ void stage32(u16* sxh, const u16* __restrict__ xsrc,
                                        const u16* __restrict__ hpart,
                                        const u16* __restrict__ hmean,
                                        const u16* __restrict__ jetsb,
                                        int b, int n0, int lane) {
    int r = lane;  // one row per lane
    const uint4* msrc = (const uint4*)(xsrc + ((size_t)(b * NN + n0 + r)) * HH);
    const uint4* hmsr = (const uint4*)(hmean + (size_t)b * HH);
    const uint2* hsrc = (const uint2*)(hpart + ((size_t)(b * NN + n0 + r)) * HH);
    u16* row = sxh + r * GINP;
#pragma unroll
    for (int i = 0; i < 32; ++i) {
        *(uint4*)(row + i * 8)       = msrc[i];      // cols   0..255 : msg
        *(uint4*)(row + 256 + i * 8) = hmsr[i];      // cols 256..511 : h_mean
    }
    *(uint4*)(row + 512) = *(const uint4*)(jetsb + (size_t)b * FJ);  // cols 512..519
#pragma unroll
    for (int i = 0; i < 64; ++i)                     // cols 520..775 : h (or r*h)
        *(uint2*)(row + 520 + i * 4) = hsrc[i];
    uint2 zz = make_uint2(0u, 0u);
#pragma unroll
    for (int i = 0; i < 6; ++i) *(uint2*)(row + 776 + i * 4) = zz;   // K pad
}

// ---------------- K3a: z and r gates; write z (bf16) and r*h (bf16) ----------------
__global__ __launch_bounds__(32)
void k3a_gates(const u16* __restrict__ msgbf, const u16* __restrict__ hbf,
               const u16* __restrict__ hmean, const u16* __restrict__ jetsb,
               const float* __restrict__ h,
               const u16* __restrict__ WzT, const u16* __restrict__ WrT,
               const float* __restrict__ bz, const float* __restrict__ br,
               u16* __restrict__ zbuf, u16* __restrict__ rhbuf) {
    __shared__ u16 sxh[32 * GINP];   // 51.2 KB
    int lane = threadIdx.x, blk = blockIdx.x;
    int b = blk >> 4, n0 = (blk & 15) << 5;
    stage32(sxh, msgbf, hbf, hmean, jetsb, b, n0, lane);
    __syncthreads();
    int m = lane & 15, half = lane >> 4, cl = lane & 15;
    for (int c2 = 0; c2 < 16; ++c2) {
        int col = c2 * 16 + cl;
        const u16* wz = WzT + (size_t)col * GINP;
        const u16* wr = WrT + (size_t)col * GINP;
        __builtin_prefetch(wz + 16 * GINP, 0, 1);
        __builtin_prefetch(wr + 16 * GINP, 0, 1);
        v8f z0 = {}, z1 = {}, r0 = {}, r1 = {};
        for (int kc = 0; kc < 25; ++kc) {
            FragBF a0, a1, bz_, br_;
            const u16* p0 = sxh + m * GINP + kc * 32 + half * 8;
            const u16* p1 = sxh + (16 + m) * GINP + kc * 32 + half * 8;
            a0.q[0] = *(const uint4*)p0; a0.q[1] = *(const uint4*)(p0 + 16);
            a1.q[0] = *(const uint4*)p1; a1.q[1] = *(const uint4*)(p1 + 16);
            const u16* pz = wz + kc * 32 + half * 8;
            const u16* pr = wr + kc * 32 + half * 8;
            bz_.q[0] = *(const uint4*)pz; bz_.q[1] = *(const uint4*)(pz + 16);
            br_.q[0] = *(const uint4*)pr; br_.q[1] = *(const uint4*)(pr + 16);
            z0 = wmma_bf(a0, bz_, z0); z1 = wmma_bf(a1, bz_, z1);
            r0 = wmma_bf(a0, br_, r0); r1 = wmma_bf(a1, br_, r1);
        }
        float bzc = bz[col], brc = br[col];
#pragma unroll
        for (int v = 0; v < 8; ++v) {
            int g0 = n0 + v + 8 * half, g1 = g0 + 16;
            float zz0 = 1.f / (1.f + __expf(-(z0[v] + bzc)));
            float zz1 = 1.f / (1.f + __expf(-(z1[v] + bzc)));
            zbuf[((size_t)(b * NN + g0)) * HH + col] = f2bf(zz0);
            zbuf[((size_t)(b * NN + g1)) * HH + col] = f2bf(zz1);
            float rv0 = 1.f / (1.f + __expf(-(r0[v] + brc)));
            float rv1 = 1.f / (1.f + __expf(-(r1[v] + brc)));
            float hv0 = h[((size_t)(b * NN + g0)) * HH + col];
            float hv1 = h[((size_t)(b * NN + g1)) * HH + col];
            rhbuf[((size_t)(b * NN + g0)) * HH + col] = f2bf(rv0 * hv0);
            rhbuf[((size_t)(b * NN + g1)) * HH + col] = f2bf(rv1 * hv1);
        }
    }
}

// ---------------- K3b: h_tilde GEMM + GRU combine ----------------
__global__ __launch_bounds__(32)
void k3b_out(const u16* __restrict__ msgbf, const u16* __restrict__ rhbuf,
             const u16* __restrict__ hmean, const u16* __restrict__ jetsb,
             const float* __restrict__ h,
             const u16* __restrict__ WhT, const float* __restrict__ bh,
             const u16* __restrict__ zbuf, float* __restrict__ out) {
    __shared__ u16 sxh[32 * GINP];   // 51.2 KB
    int lane = threadIdx.x, blk = blockIdx.x;
    int b = blk >> 4, n0 = (blk & 15) << 5;
    stage32(sxh, msgbf, rhbuf, hmean, jetsb, b, n0, lane);
    __syncthreads();
    int m = lane & 15, half = lane >> 4, cl = lane & 15;
    for (int c2 = 0; c2 < 16; ++c2) {
        int col = c2 * 16 + cl;
        const u16* wh = WhT + (size_t)col * GINP;
        __builtin_prefetch(wh + 16 * GINP, 0, 1);
        v8f h0 = {}, h1 = {};
        for (int kc = 0; kc < 25; ++kc) {
            FragBF a0, a1, bb;
            const u16* p0 = sxh + m * GINP + kc * 32 + half * 8;
            const u16* p1 = sxh + (16 + m) * GINP + kc * 32 + half * 8;
            a0.q[0] = *(const uint4*)p0; a0.q[1] = *(const uint4*)(p0 + 16);
            a1.q[0] = *(const uint4*)p1; a1.q[1] = *(const uint4*)(p1 + 16);
            const u16* p = wh + kc * 32 + half * 8;
            bb.q[0] = *(const uint4*)p; bb.q[1] = *(const uint4*)(p + 16);
            h0 = wmma_bf(a0, bb, h0);
            h1 = wmma_bf(a1, bb, h1);
        }
        float bhc = bh[col];
#pragma unroll
        for (int v = 0; v < 8; ++v) {
            int g0 = n0 + v + 8 * half, g1 = g0 + 16;
            float ht0 = tanhf(h0[v] + bhc), ht1 = tanhf(h1[v] + bhc);
            float zv0 = bf2f(zbuf[((size_t)(b * NN + g0)) * HH + col]);
            float zv1 = bf2f(zbuf[((size_t)(b * NN + g1)) * HH + col]);
            float hv0 = h[((size_t)(b * NN + g0)) * HH + col];
            float hv1 = h[((size_t)(b * NN + g1)) * HH + col];
            out[((size_t)(b * NN + g0)) * HH + col] = (1.f - zv0) * hv0 + zv0 * ht0;
            out[((size_t)(b * NN + g1)) * HH + col] = (1.f - zv1) * hv1 + zv1 * ht1;
        }
    }
}

// ---------------- launcher ----------------
extern "C" void kernel_launch(void* const* d_in, const int* in_sizes, int n_in,
                              void* d_out, int out_size, void* d_ws, size_t ws_size,
                              hipStream_t stream) {
    (void)in_sizes; (void)n_in; (void)out_size; (void)ws_size;
    const float* h    = (const float*)d_in[0];
    const float* jets = (const float*)d_in[1];
    const float* mask = (const float*)d_in[2];
    const float* Wa   = (const float*)d_in[3];
    const float* ba   = (const float*)d_in[4];
    const float* Wm   = (const float*)d_in[5];
    const float* bm   = (const float*)d_in[6];
    const float* Wz   = (const float*)d_in[7];
    const float* bz   = (const float*)d_in[8];
    const float* Wr   = (const float*)d_in[9];
    const float* br   = (const float*)d_in[10];
    const float* Wh   = (const float*)d_in[11];
    const float* bh   = (const float*)d_in[12];

    char* ws = (char*)d_ws;
    u16* hbf   = (u16*)(ws + OFF_HBF);
    u16* qbf   = (u16*)(ws + OFF_QBF);    // K3a reuses as r*h
    u16* mTbf  = (u16*)(ws + OFF_MT);     // K3a reuses as z
    u16* msgbf = (u16*)(ws + OFF_MSG);
    u16* WaT   = (u16*)(ws + OFF_WAT);
    u16* WmT   = (u16*)(ws + OFF_WMT);
    u16* WzT   = (u16*)(ws + OFF_WZT);
    u16* WrT   = (u16*)(ws + OFF_WRT);
    u16* WhT   = (u16*)(ws + OFF_WHT);
    u16* hmean = (u16*)(ws + OFF_HMEAN);
    u16* jetsb = (u16*)(ws + OFF_JETS);

    k_prep_hh<<<HH, 256, 0, stream>>>(Wa, WaT);
    k_prep_hh<<<HH, 256, 0, stream>>>(Wm, WmT);
    k_prep_gin<<<HH, 256, 0, stream>>>(Wz, WzT);
    k_prep_gin<<<HH, 256, 0, stream>>>(Wr, WrT);
    k_prep_gin<<<HH, 256, 0, stream>>>(Wh, WhT);
    k_mean_jets<<<BB, 256, 0, stream>>>(h, jets, hmean, jetsb);

    int tiles = BB * (NN / 32);   // 2048 (32 rows per wave)
    k1_qm<<<tiles, 32, 0, stream>>>(h, ba, bm, WaT, WmT, hbf, qbf, mTbf);
    k2_attn<<<tiles, 32, 0, stream>>>(qbf, hbf, mTbf, mask, msgbf);
    k3a_gates<<<tiles, 32, 0, stream>>>(msgbf, hbf, hmean, jetsb, h,
                                        WzT, WrT, bz, br, mTbf /*z*/, qbf /*r*h*/);
    k3b_out<<<tiles, 32, 0, stream>>>(msgbf, qbf /*r*h*/, hmean, jetsb, h,
                                      WhT, bh, mTbf /*z*/, (float*)d_out);
}
